// GraphTransformer_47880295416408
// MI455X (gfx1250) — matrix operand order repaired and checked by reading, not compiled, not run on hardware
//
#include <hip/hip_runtime.h>
#include <math.h>

#define NNODES 50000
#define NEDGES 800000
#define INC    128
#define OUTC   16
#define NHEADS 8
#define HD     128      // NHEADS*OUTC
#define WCOLS  400      // 3*HD + OUTC  (Q|K|V|skip fused)
#define NTILES 5        // column tiles per wave (5*16 = 80 cols; 400/80 = 5 panels)
#define MT     3125     // NNODES/16 row tiles
#define WAVES_PER_BLK 4

typedef __attribute__((ext_vector_type(16))) _Float16 v16h;
typedef __attribute__((ext_vector_type(8)))  _Float16 v8h;
typedef __attribute__((ext_vector_type(8)))  float    v8f;

// ---------- monotonic float <-> uint encoding for atomicMax on f32 ----------
__device__ __forceinline__ unsigned fenc(float f) {
    unsigned u = __float_as_uint(f);
    return (u & 0x80000000u) ? ~u : (u | 0x80000000u);
}
__device__ __forceinline__ float fdec(unsigned u) {
    unsigned b = (u & 0x80000000u) ? (u & 0x7fffffffu) : ~u;
    return __uint_as_float(b);
}
#define ENC_NEG_INF 0x007FFFFFu   // fenc(-inf)

// ---------- casts / weight fusion ----------
__global__ void cast_x_kernel(const float* __restrict__ x, _Float16* __restrict__ x16) {
    int t = blockIdx.x * blockDim.x + threadIdx.x;
    if (t < NNODES * INC) x16[t] = (_Float16)x[t];
}

__global__ void build_w_kernel(const float* __restrict__ Wq, const float* __restrict__ Wk,
                               const float* __restrict__ Wv, const float* __restrict__ Ws,
                               const float* __restrict__ bq, const float* __restrict__ bk,
                               const float* __restrict__ bv, const float* __restrict__ bs,
                               _Float16* __restrict__ wt16, float* __restrict__ ball) {
    int t = blockIdx.x * blockDim.x + threadIdx.x;
    if (t >= WCOLS * INC) return;
    int j = t / INC;     // fused output column 0..399
    int k = t % INC;     // input channel
    float w;
    if      (j <   HD) w = Wq[k * HD + j];
    else if (j < 2*HD) w = Wk[k * HD + (j - HD)];
    else if (j < 3*HD) w = Wv[k * HD + (j - 2*HD)];
    else               w = Ws[k * OUTC + (j - 3*HD)];
    wt16[(size_t)j * INC + k] = (_Float16)w;   // column-major (transposed) for WMMA B frag
    if (k == 0) {
        float b;
        if      (j <   HD) b = bq[j];
        else if (j < 2*HD) b = bk[j - HD];
        else if (j < 3*HD) b = bv[j - 2*HD];
        else               b = bs[j - 3*HD];
        ball[j] = b;
    }
}

// ---------- fused QKVS GEMM ----------
// Each wave computes a 16x80 output panel (5 x 16x16 tiles): A fragment loaded
// once per K-chunk and reused across 5 WMMAs -> 20 v_wmma per wave, 5x less A
// traffic than one-tile-per-wave. 4 waves per block cover 4 consecutive M tiles.
__global__ __launch_bounds__(32 * WAVES_PER_BLK) void qkvs_gemm_kernel(
        const _Float16* __restrict__ x16,
        const _Float16* __restrict__ wt16,
        const float* __restrict__ ball,
        float* __restrict__ qkvs) {
    const int lane = threadIdx.x & 31;
    const int wave = threadIdx.x >> 5;
    const int mt   = blockIdx.y * WAVES_PER_BLK + wave;   // row tile index
    if (mt >= MT) return;                                  // wave-uniform: EXEC stays all-ones
    const int m0 = mt * 16;
    const int n0 = blockIdx.x * (NTILES * 16);             // 80-column panel base

    const int lo  = lane & 15;
    const int hi  = lane >> 4;          // 0 or 1 (lane half)
    const int row = m0 + lo;            // A: lane -> matrix row

    const _Float16* arow = x16 + (size_t)row * INC;
    const _Float16* bcol[NTILES];
    #pragma unroll
    for (int j = 0; j < NTILES; ++j)
        bcol[j] = wt16 + (size_t)(n0 + j * 16 + lo) * INC;  // B/D: lane -> matrix column

    v8f acc[NTILES];
    #pragma unroll
    for (int j = 0; j < NTILES; ++j) acc[j] = (v8f){};

    #pragma unroll
    for (int kc = 0; kc < INC; kc += 32) {
        // A 16x32 f16 frag: lane-half selects K 0-15 vs 16-31
        v16h a = *(const v16h*)(arow + kc + hi * 16);
        #pragma unroll
        for (int j = 0; j < NTILES; ++j) {
            // B 32x16 f16 frag: lane = column, lane-half selects K 0-15 vs 16-31
            v16h b = *(const v16h*)(bcol[j] + kc + hi * 16);
            acc[j] = __builtin_amdgcn_wmma_f32_16x16x32_f16(
                         /*neg_a=*/false, a, /*neg_b=*/false, b,
                         /*c_mod=*/(short)0, acc[j], /*reuse_a=*/false, /*reuse_b=*/false);
        }
    }
    #pragma unroll
    for (int j = 0; j < NTILES; ++j) {
        const int col = n0 + j * 16 + lo;
        const float bb = ball[col];
        #pragma unroll
        for (int r = 0; r < 8; ++r) {
            int m = m0 + r + hi * 8;    // D layout: VGPR r -> row r (lanes 0-15) / r+8 (lanes 16-31)
            qkvs[(size_t)m * WCOLS + col] = acc[j][r] + bb;
        }
    }
}

// ---------- scatter-softmax state init ----------
__global__ void init_kernel(unsigned* __restrict__ segmax, float* __restrict__ denom,
                            float* __restrict__ agg) {
    int t = blockIdx.x * blockDim.x + threadIdx.x;
    if (t < NNODES * NHEADS) { segmax[t] = ENC_NEG_INF; denom[t] = 0.0f; }
    if (t < NNODES * NHEADS * OUTC) agg[t] = 0.0f;
}

// ---------- pass 1: per-(edge,head) logits + segment max ----------
__global__ void edge_logits_kernel(const long long* __restrict__ ei,
                                   const float* __restrict__ qkvs,
                                   float* __restrict__ logits,
                                   unsigned* __restrict__ segmax) {
    int t = blockIdx.x * blockDim.x + threadIdx.x;
    if (t >= NEDGES * NHEADS) return;
    int e = t >> 3, h = t & 7;
    int src = (int)ei[e];
    int dst = (int)ei[NEDGES + e];
    const float4* q = (const float4*)(qkvs + (size_t)dst * WCOLS + h * OUTC);        // Q block
    const float4* k = (const float4*)(qkvs + (size_t)src * WCOLS + HD + h * OUTC);   // K block
    float s = 0.0f;
    #pragma unroll
    for (int i = 0; i < 4; ++i) {
        float4 qa = q[i], ka = k[i];
        s += qa.x * ka.x + qa.y * ka.y + qa.z * ka.z + qa.w * ka.w;
    }
    s *= 0.25f;                      // 1/sqrt(OUTC)
    logits[t] = s;
    atomicMax(&segmax[dst * NHEADS + h], fenc(s));
}

// ---------- pass 2: exp(logit - max), segment sum ----------
__global__ void edge_exp_kernel(const long long* __restrict__ ei,
                                const unsigned* __restrict__ segmax,
                                float* __restrict__ logits,
                                float* __restrict__ denom) {
    int t = blockIdx.x * blockDim.x + threadIdx.x;
    if (t >= NEDGES * NHEADS) return;
    int e = t >> 3, h = t & 7;
    int dst = (int)ei[NEDGES + e];
    float m  = fdec(segmax[dst * NHEADS + h]);   // finite: this dst has >=1 edge
    float ex = __expf(logits[t] - m);
    logits[t] = ex;
    atomicAdd(&denom[dst * NHEADS + h], ex);
}

// ---------- pass 3: alpha * v scatter-add ----------
__global__ void edge_scatter_kernel(const long long* __restrict__ ei,
                                    const float* __restrict__ qkvs,
                                    const float* __restrict__ logits,
                                    const float* __restrict__ denom,
                                    float* __restrict__ agg) {
    int t = blockIdx.x * blockDim.x + threadIdx.x;
    if (t >= NEDGES * NHEADS) return;
    int e = t >> 3, h = t & 7;
    int src = (int)ei[e];
    int dst = (int)ei[NEDGES + e];
    float d = denom[dst * NHEADS + h];
    float alpha = logits[t] / fmaxf(d, 1e-16f);
    const float* v = qkvs + (size_t)src * WCOLS + 2 * HD + h * OUTC;   // V block
    float* a = agg + ((size_t)dst * NHEADS + h) * OUTC;
    #pragma unroll
    for (int i = 0; i < OUTC; ++i) atomicAdd(&a[i], alpha * v[i]);
}

// ---------- pass 4: head mean + skip ----------
__global__ void finalize_kernel(const float* __restrict__ qkvs,
                                const float* __restrict__ agg,
                                float* __restrict__ out) {
    int t = blockIdx.x * blockDim.x + threadIdx.x;
    if (t >= NNODES * OUTC) return;
    int n = t / OUTC, d = t % OUTC;
    float s = 0.0f;
    #pragma unroll
    for (int h = 0; h < NHEADS; ++h) s += agg[((size_t)n * NHEADS + h) * OUTC + d];
    out[t] = s * (1.0f / NHEADS) + qkvs[(size_t)n * WCOLS + 3 * HD + d];   // skip (bias fused)
}

static inline size_t alignUp(size_t x) { return (x + 255) & ~(size_t)255; }

extern "C" void kernel_launch(void* const* d_in, const int* in_sizes, int n_in,
                              void* d_out, int out_size, void* d_ws, size_t ws_size,
                              hipStream_t stream) {
    (void)in_sizes; (void)n_in; (void)out_size; (void)ws_size;
    const float*     x  = (const float*)d_in[0];
    const long long* ei = (const long long*)d_in[1];   // int64 edge_index (2,E)
    const float* Wq = (const float*)d_in[2];
    const float* bq = (const float*)d_in[3];
    const float* Wk = (const float*)d_in[4];
    const float* bk = (const float*)d_in[5];
    const float* Wv = (const float*)d_in[6];
    const float* bv = (const float*)d_in[7];
    const float* Ws = (const float*)d_in[8];
    const float* bs = (const float*)d_in[9];
    float* out = (float*)d_out;

    // workspace carve-out (256B aligned regions)
    char* ws = (char*)d_ws;
    size_t off = 0;
    _Float16* x16  = (_Float16*)(ws + off); off = alignUp(off + (size_t)NNODES * INC * 2);
    _Float16* wt16 = (_Float16*)(ws + off); off = alignUp(off + (size_t)WCOLS * INC * 2);
    float*    ball = (float*)   (ws + off); off = alignUp(off + (size_t)WCOLS * 4);
    float*    qkvs = (float*)   (ws + off); off = alignUp(off + (size_t)NNODES * WCOLS * 4);
    float*    lgt  = (float*)   (ws + off); off = alignUp(off + (size_t)NEDGES * NHEADS * 4);
    unsigned* smax = (unsigned*)(ws + off); off = alignUp(off + (size_t)NNODES * NHEADS * 4);
    float*    den  = (float*)   (ws + off); off = alignUp(off + (size_t)NNODES * NHEADS * 4);
    float*    agg  = (float*)   (ws + off); off = alignUp(off + (size_t)NNODES * NHEADS * OUTC * 4);

    const int TB = 256;
    cast_x_kernel<<<(NNODES * INC + TB - 1) / TB, TB, 0, stream>>>(x, x16);
    build_w_kernel<<<(WCOLS * INC + TB - 1) / TB, TB, 0, stream>>>(Wq, Wk, Wv, Ws, bq, bk, bv, bs, wt16, ball);
    init_kernel<<<(NNODES * NHEADS * OUTC + TB - 1) / TB, TB, 0, stream>>>(smax, den, agg);

    // 5 column panels x ceil(3125/4) row-tile groups, 4 waves (128 threads) per block
    qkvs_gemm_kernel<<<dim3(WCOLS / (NTILES * 16), (MT + WAVES_PER_BLK - 1) / WAVES_PER_BLK),
                       32 * WAVES_PER_BLK, 0, stream>>>(x16, wt16, ball, qkvs);

    const int EH = NEDGES * NHEADS;
    edge_logits_kernel <<<(EH + TB - 1) / TB, TB, 0, stream>>>(ei, qkvs, lgt, smax);
    edge_exp_kernel    <<<(EH + TB - 1) / TB, TB, 0, stream>>>(ei, smax, lgt, den);
    edge_scatter_kernel<<<(EH + TB - 1) / TB, TB, 0, stream>>>(ei, qkvs, lgt, den, agg);
    finalize_kernel    <<<(NNODES * OUTC + TB - 1) / TB, TB, 0, stream>>>(qkvs, agg, out);
}
